// ProvenanceGNN_37297495998502
// MI455X (gfx1250) — matrix operand order repaired and checked by reading, not compiled, not run on hardware
//
#include <hip/hip_runtime.h>
#include <hip/hip_bf16.h>
#include <math.h>

// ---------------- problem constants (from reference) ----------------
static constexpr int NN      = 100000;  // nodes
static constexpr int EE      = 800000;  // edges (before self loops)
static constexpr int ET      = EE + NN; // edges incl self loops
static constexpr int IN_DIM  = 128;
static constexpr int HID     = 128;
static constexpr int HEADS   = 4;
static constexpr int CH      = 32;      // HID/HEADS
static constexpr int TD      = 16;
static constexpr int KPROJ   = 160;     // 144 padded to mult of 32
static constexpr int OUTD    = 2;

typedef __bf16 bf16;
typedef __attribute__((ext_vector_type(16))) bf16  v16bf;
typedef __attribute__((ext_vector_type(8)))  bf16  v8bf;
typedef __attribute__((ext_vector_type(8)))  float v8f;

// ---------------- helpers ----------------
__device__ __forceinline__ unsigned enc_ord(float f) {
  unsigned b = __float_as_uint(f);
  return (b & 0x80000000u) ? ~b : (b | 0x80000000u);
}
__device__ __forceinline__ float dec_ord(unsigned u) {
  return (u & 0x80000000u) ? __uint_as_float(u & 0x7FFFFFFFu)
                           : __uint_as_float(~u);
}
__device__ __forceinline__ void edge_sd(const int* __restrict__ ei, int e, int& s, int& d) {
  if (e < EE) { s = ei[e]; d = ei[EE + e]; }
  else        { s = d = e - EE; }
}

// ---------------- min/max of timesteps ----------------
__global__ void init_minmax(int* mm) {
  if (blockIdx.x == 0 && threadIdx.x == 0) { mm[0] = 0x7FFFFFFF; mm[1] = -0x7FFFFFFF - 1; }
}
__global__ void reduce_minmax(const int* __restrict__ ts, int n, int* mm) {
  __shared__ int smn[256], smx[256];
  int i = blockIdx.x * blockDim.x + threadIdx.x;
  int v = (i < n) ? ts[i] : ts[0];
  smn[threadIdx.x] = v; smx[threadIdx.x] = v;
  __syncthreads();
  for (int s = 128; s > 0; s >>= 1) {
    if ((int)threadIdx.x < s) {
      smn[threadIdx.x] = min(smn[threadIdx.x], smn[threadIdx.x + s]);
      smx[threadIdx.x] = max(smx[threadIdx.x], smx[threadIdx.x + s]);
    }
    __syncthreads();
  }
  if (threadIdx.x == 0) { atomicMin(&mm[0], smn[0]); atomicMax(&mm[1], smx[0]); }
}

// ---------------- weight prep: bf16 + WMMA-lane swizzle ----------------
// Swizzled layout: elem[((kt*NT + nt)*32 + lane)*16 + j] = B[kt*32 + (lane/16)*16 + j][nt*16 + lane%16]
__device__ __forceinline__ void swz_one(int t, const float* __restrict__ src,
                                        int realK, int ncols, int NT,
                                        bf16* __restrict__ dst) {
  int j    = t & 15;
  int lane = (t >> 4) & 31;
  int rest = t >> 9;
  int nt   = rest % NT;
  int kt   = rest / NT;
  int K    = kt * 32 + (lane >> 4) * 16 + j;
  int col  = nt * 16 + (lane & 15);
  float v  = (K < realK) ? src[(size_t)K * ncols + col] : 0.0f;
  dst[t] = (bf16)v;
}

__global__ void prep_weights(const float* __restrict__ proj_w,
                             const float* __restrict__ conv_w,
                             const float* __restrict__ head_w1,
                             const float* __restrict__ bn_gamma,
                             const float* __restrict__ bn_var,
                             bf16* __restrict__ wswz,
                             float* __restrict__ bnscale) {
  int tid = blockIdx.x * blockDim.x + threadIdx.x;
  // segments: proj 5*8*512=20480 | conv0 16384 | conv1 16384 | head1 8192 | bnscale 256
  if (tid < 20480)            swz_one(tid,          proj_w,            144, 128, 8, wswz);
  else if (tid < 36864)       swz_one(tid - 20480,  conv_w,            128, 128, 8, wswz + 20480);
  else if (tid < 53248)       swz_one(tid - 36864,  conv_w + 128*128,  128, 128, 8, wswz + 36864);
  else if (tid < 61440)       swz_one(tid - 53248,  head_w1,           128,  64, 4, wswz + 53248);
  else if (tid < 61696) {
    int c = tid - 61440; // 0..255 => [layer*128 + ch]
    bnscale[c] = bn_gamma[c] * rsqrtf(bn_var[c] + 1e-5f);
  }
}

// ---------------- per-node prep: concat(x,te) staging (bf16) + gate ----------------
__global__ void node_prep(const float* __restrict__ x, const int* __restrict__ ts,
                          const int* __restrict__ mm,
                          const float* __restrict__ gw1, const float* __restrict__ gb1,
                          const float* __restrict__ gw2, const float* __restrict__ gb2,
                          bf16* __restrict__ Abf, float* __restrict__ gate) {
  int n = blockIdx.x * blockDim.x + threadIdx.x;
  if (n >= NN) return;
  float t = (float)ts[n];
  bf16* row = Abf + (size_t)n * KPROJ;
  const float* xr = x + (size_t)n * IN_DIM;
  #pragma unroll 4
  for (int i = 0; i < IN_DIM; ++i) row[i] = (bf16)xr[i];
  const float c = -logf(10000.0f) / (float)TD;
  #pragma unroll
  for (int i = 0; i < 8; ++i) {
    float fr = expf((2.0f * i) * c);
    float a  = t * fr;
    row[IN_DIM + i]     = (bf16)sinf(a);
    row[IN_DIM + 8 + i] = (bf16)cosf(a);
  }
  #pragma unroll
  for (int i = IN_DIM + TD; i < KPROJ; ++i) row[i] = (bf16)0.0f;
  // age gate: 1 -> 16 -> 1 MLP + sigmoid
  float mn = (float)mm[0], mx = (float)mm[1];
  float age = (t - mn) / fmaxf(mx - mn, 1.0f);
  float s = gb2[0];
  #pragma unroll
  for (int j = 0; j < 16; ++j) {
    float v = age * gw1[j] + gb1[j];
    v = v > 0.0f ? v : 0.0f;
    s += v * gw2[j];
  }
  gate[n] = 1.0f / (1.0f + expf(-s));
}

// ---------------- bf16 WMMA GEMM: one wave per 16-row x (NT*16)-col stripe ----------------
// Per K-step: A fragment loaded once, all NT B fragments issued as one load clause,
// then NT back-to-back WMMAs drain the loads with partial waits (latency overlap).
// A: row-major [M x K] bf16 (lda), B: pre-swizzled, K mult of 32, M mult of 16.
template <int NT, int KTILES>
__global__ void __launch_bounds__(32)
gemm_bf16_wmma(const bf16* __restrict__ A, int lda,
               const bf16* __restrict__ Bswz,
               const float* __restrict__ bias,
               float* __restrict__ Cout, int ldc,
               bf16* __restrict__ Cbf, int ldcb,
               int act) {
  const int lane = threadIdx.x & 31;
  const int half = lane >> 4;
  const int mrow = lane & 15;
  const int row0 = blockIdx.x * 16;

  v8f acc[NT];
  #pragma unroll
  for (int nt = 0; nt < NT; ++nt) acc[nt] = (v8f){};

  const bf16* arow = A + (size_t)(row0 + mrow) * lda;
  const bf16* bptr = Bswz + (size_t)lane * 16;   // + ((kt*NT + nt)*512)

  #pragma unroll
  for (int kt = 0; kt < KTILES; ++kt) {
    const int k0 = kt << 5;
    // A fragment: contiguous 16-elem k chunk per lane (k = 16*half + s), matching B swizzle
    v8bf alo = *(const v8bf*)(arow + k0 + 16 * half);
    v8bf ahi = *(const v8bf*)(arow + k0 + 16 * half + 8);
    v16bf a = __builtin_shufflevector(alo, ahi, 0,1,2,3,4,5,6,7,8,9,10,11,12,13,14,15);
    // issue ALL B-fragment loads for this k-step first (one clause),
    // then the WMMA group drains them with partial loadcnt waits
    v16bf b[NT];
    #pragma unroll
    for (int nt = 0; nt < NT; ++nt)
      b[nt] = *(const v16bf*)(bptr + ((size_t)kt * NT + nt) * 512);
    #pragma unroll
    for (int nt = 0; nt < NT; ++nt)
      acc[nt] = __builtin_amdgcn_wmma_f32_16x16x32_bf16(false, a, false, b[nt],
                                                        (short)0, acc[nt], false, false);
  }
  // epilogue: element r -> M = r + 8*half, N = lane%16
  #pragma unroll
  for (int nt = 0; nt < NT; ++nt) {
    const int col = nt * 16 + mrow;
    const float bv = bias ? bias[col] : 0.0f;
    #pragma unroll
    for (int r = 0; r < 8; ++r) {
      int m = row0 + r + 8 * half;
      float v = acc[nt][r] + bv;
      if (act == 1) v = v > 0.0f ? v : 0.0f;
      if (Cout) Cout[(size_t)m * ldc + col] = v;
      if (Cbf)  Cbf[(size_t)m * ldcb + col] = (bf16)v;
    }
  }
}

// ---------------- per-(node,head) attention logits ----------------
__global__ void att_node(const float* __restrict__ ht,
                         const float* __restrict__ asrc, const float* __restrict__ adst,
                         float* __restrict__ al_s, float* __restrict__ al_d) {
  int tid = blockIdx.x * blockDim.x + threadIdx.x;
  if (tid >= NN * HEADS) return;
  int n = tid >> 2, hd = tid & 3;
  const float* hrow = ht + (size_t)n * HID + hd * CH;
  float ss = 0.0f, sd = 0.0f;
  #pragma unroll 8
  for (int k = 0; k < CH; ++k) {
    float v = hrow[k];
    ss += v * asrc[hd * CH + k];
    sd += v * adst[hd * CH + k];
  }
  al_s[tid] = ss; al_d[tid] = sd;
}

// zero agg (N*HID) and init per-(node,head) softmax state in one pass
__global__ void init_layer(float* __restrict__ agg,
                           unsigned* __restrict__ mEnc, float* __restrict__ den) {
  int i = blockIdx.x * blockDim.x + threadIdx.x;
  if (i < NN * HID) agg[i] = 0.0f;
  if (i < NN * HEADS) { mEnc[i] = 0u; den[i] = 0.0f; }
}

// ---------------- edge passes (softmax over dst segments) ----------------
__global__ void edge_max(const int* __restrict__ ei,
                         const float* __restrict__ al_s, const float* __restrict__ al_d,
                         unsigned* __restrict__ mEnc) {
  int tid = blockIdx.x * blockDim.x + threadIdx.x;
  if (tid >= ET * HEADS) return;
  int e = tid >> 2, hd = tid & 3;
  int s, d; edge_sd(ei, e, s, d);
  float a = al_s[s * HEADS + hd] + al_d[d * HEADS + hd];
  a = a > 0.0f ? a : 0.2f * a;
  atomicMax(&mEnc[d * HEADS + hd], enc_ord(a));
}

__global__ void edge_den(const int* __restrict__ ei,
                         const float* __restrict__ al_s, const float* __restrict__ al_d,
                         const unsigned* __restrict__ mEnc, float* __restrict__ den) {
  int tid = blockIdx.x * blockDim.x + threadIdx.x;
  if (tid >= ET * HEADS) return;
  int e = tid >> 2, hd = tid & 3;
  int s, d; edge_sd(ei, e, s, d);
  float a = al_s[s * HEADS + hd] + al_d[d * HEADS + hd];
  a = a > 0.0f ? a : 0.2f * a;
  float m = dec_ord(mEnc[d * HEADS + hd]);
  atomicAdd(&den[d * HEADS + hd], __expf(a - m));
}

__global__ void edge_agg(const int* __restrict__ ei,
                         const float* __restrict__ al_s, const float* __restrict__ al_d,
                         const unsigned* __restrict__ mEnc, const float* __restrict__ den,
                         const float* __restrict__ ht, float* __restrict__ agg) {
  int tid = blockIdx.x * blockDim.x + threadIdx.x;  // ET*32 = 28.8M < 2^31
  if (tid >= ET * 32) return;
  int e = tid >> 5;
  int t = tid & 31;
  int hd = t >> 3;        // 8 threads per head, 4 feats each
  int f0 = t * 4;
  int s, d; edge_sd(ei, e, s, d);
  float a = al_s[s * HEADS + hd] + al_d[d * HEADS + hd];
  a = a > 0.0f ? a : 0.2f * a;
  float m  = dec_ord(mEnc[d * HEADS + hd]);
  float at = __expf(a - m) / (den[d * HEADS + hd] + 1e-16f);
  const float4 hv = *(const float4*)(ht + (size_t)s * HID + f0);
  float* ap = agg + (size_t)d * HID + f0;
  atomicAdd(ap + 0, hv.x * at);
  atomicAdd(ap + 1, hv.y * at);
  atomicAdd(ap + 2, hv.z * at);
  atomicAdd(ap + 3, hv.w * at);
}

// ---------------- bias + BN + ELU + gate (in-place) -> bf16 staging ----------------
__global__ void post_layer(float* __restrict__ agg,
                           const float* __restrict__ conv_b,
                           const float* __restrict__ bn_mean,
                           const float* __restrict__ bn_beta,
                           const float* __restrict__ bnscale,
                           const float* __restrict__ gate,
                           bf16* __restrict__ Hbf) {
  int tid = blockIdx.x * blockDim.x + threadIdx.x;
  if (tid >= NN * HID) return;
  int n = tid >> 7, c = tid & 127;
  float v = agg[tid] + conv_b[c];
  v = (v - bn_mean[c]) * bnscale[c] + bn_beta[c];
  v = v > 0.0f ? v : (__expf(v) - 1.0f);
  v *= gate[n];
  agg[tid] = v;
  Hbf[tid] = (bf16)v;
}

// ---------------- final 64 -> 2 projection ----------------
__global__ void head_out(const float* __restrict__ hid,
                         const float* __restrict__ w2, const float* __restrict__ b2,
                         float* __restrict__ out) {
  int n = blockIdx.x * blockDim.x + threadIdx.x;
  if (n >= NN) return;
  const float* r = hid + (size_t)n * 64;
  float o0 = b2[0], o1 = b2[1];
  #pragma unroll 8
  for (int k = 0; k < 64; ++k) {
    float v = r[k];
    o0 += v * w2[k * 2 + 0];
    o1 += v * w2[k * 2 + 1];
  }
  out[n * OUTD + 0] = o0;
  out[n * OUTD + 1] = o1;
}

// ---------------- host side ----------------
extern "C" void kernel_launch(void* const* d_in, const int* in_sizes, int n_in,
                              void* d_out, int out_size, void* d_ws, size_t ws_size,
                              hipStream_t stream) {
  const float* x        = (const float*)d_in[0];
  const int*   ei       = (const int*)  d_in[1];
  const int*   ts       = (const int*)  d_in[2];
  const float* proj_w   = (const float*)d_in[3];
  const float* proj_b   = (const float*)d_in[4];
  const float* conv_w   = (const float*)d_in[5];
  const float* conv_b   = (const float*)d_in[6];
  const float* att_src  = (const float*)d_in[7];
  const float* att_dst  = (const float*)d_in[8];
  const float* bn_gamma = (const float*)d_in[9];
  const float* bn_beta  = (const float*)d_in[10];
  const float* bn_mean  = (const float*)d_in[11];
  const float* bn_var   = (const float*)d_in[12];
  const float* gate_w1  = (const float*)d_in[13];
  const float* gate_b1  = (const float*)d_in[14];
  const float* gate_w2  = (const float*)d_in[15];
  const float* gate_b2  = (const float*)d_in[16];
  const float* head_w1  = (const float*)d_in[17];
  const float* head_b1  = (const float*)d_in[18];
  const float* head_w2  = (const float*)d_in[19];
  const float* head_b2  = (const float*)d_in[20];
  float* out = (float*)d_out;

  // workspace carve-out
  char* w = (char*)d_ws;
  auto carve = [&](size_t bytes) { void* p = (void*)w; w += (bytes + 255) & ~(size_t)255; return p; };
  float*    agg     = (float*)   carve((size_t)NN * HID * 4);      // aggregation / h (in-place)
  float*    T       = (float*)   carve((size_t)NN * HID * 4);      // ht = h@W ; later head hidden
  bf16*     Abf     = (bf16*)    carve((size_t)NN * KPROJ * 2);    // proj A staging
  bf16*     Hbf     = (bf16*)    carve((size_t)NN * HID * 2);      // activation staging (bf16)
  float*    al_s    = (float*)   carve((size_t)NN * HEADS * 4);
  float*    al_d    = (float*)   carve((size_t)NN * HEADS * 4);
  unsigned* mEnc    = (unsigned*)carve((size_t)NN * HEADS * 4);
  float*    den     = (float*)   carve((size_t)NN * HEADS * 4);
  float*    gate    = (float*)   carve((size_t)NN * 4);
  bf16*     wswz    = (bf16*)    carve((size_t)61440 * 2);
  float*    bnscale = (float*)   carve((size_t)256 * 4);
  int*      mm      = (int*)     carve(16);

  const int TB = 256;
  const int gN    = (NN + TB - 1) / TB;               // 391
  const int gNH   = (NN * HEADS + TB - 1) / TB;       // 1563
  const int gNF   = (NN * HID) / TB;                  // 50000
  const int gE4   = (ET * HEADS + TB - 1) / TB;       // 14063
  const int gE32  = (ET * 32) / TB;                   // 112500
  const int gW    = (61696 + TB - 1) / TB;            // 241

  init_minmax<<<1, 1, 0, stream>>>(mm);
  reduce_minmax<<<gN, TB, 0, stream>>>(ts, NN, mm);
  prep_weights<<<gW, TB, 0, stream>>>(proj_w, conv_w, head_w1, bn_gamma, bn_var, wswz, bnscale);
  node_prep<<<gN, TB, 0, stream>>>(x, ts, mm, gate_w1, gate_b1, gate_w2, gate_b2, Abf, gate);

  // projection: [N x 160] @ [160 x 128] (+bias, ReLU) -> bf16 staging Hbf
  gemm_bf16_wmma<8, 5><<<NN / 16, 32, 0, stream>>>(
      Abf, KPROJ, wswz, proj_b, (float*)nullptr, 0, Hbf, HID, 1);

  for (int l = 0; l < 2; ++l) {
    const bf16* Wl = wswz + 20480 + l * 16384;
    // ht = h @ W (no bias/act)
    gemm_bf16_wmma<8, 4><<<NN / 16, 32, 0, stream>>>(
        Hbf, HID, Wl, (const float*)nullptr, T, HID, (bf16*)nullptr, 0, 0);
    att_node<<<gNH, TB, 0, stream>>>(T, att_src + l * HID, att_dst + l * HID, al_s, al_d);
    init_layer<<<gNF, TB, 0, stream>>>(agg, mEnc, den);
    edge_max<<<gE4, TB, 0, stream>>>(ei, al_s, al_d, mEnc);
    edge_den<<<gE4, TB, 0, stream>>>(ei, al_s, al_d, mEnc, den);
    edge_agg<<<gE32, TB, 0, stream>>>(ei, al_s, al_d, mEnc, den, T, agg);
    post_layer<<<gNF, TB, 0, stream>>>(agg, conv_b + l * HID, bn_mean + l * HID,
                                       bn_beta + l * HID, bnscale + l * HID, gate, Hbf);
  }

  // head: [N x 128] @ [128 x 64] (+bias, ReLU) -> T (fp32, ld 64)
  gemm_bf16_wmma<4, 4><<<NN / 16, 32, 0, stream>>>(
      Hbf, HID, wswz + 53248, head_b1, T, 64, (bf16*)nullptr, 0, 1);
  head_out<<<gN, TB, 0, stream>>>(T, head_w2, head_b2, out);
}
